// lcc_sampling_59047210385572
// MI455X (gfx1250) — matrix operand order repaired and checked by reading, not compiled, not run on hardware
//
#include <hip/hip_runtime.h>
#include <hip/hip_bf16.h>
#include <stdint.h>

// Problem sizes (fixed by the reference).
#define Bsz 4096
#define Ssz 128
#define Asz 8192
#define Dsz 512

#define ROWS     16      // samples per workgroup
#define HALFC    4096    // columns processed per half (LDS key buffer)
#define NC       512     // candidate capacity per row
#define NTHREADS 128     // 4 waves (wave32)

#define KSTRIDE  4104    // u16 key row stride (pad vs 4096)
#define SELF_STR 516     // f32 sel row stride (pad, keeps 16B alignment: 516*4=2064=16*129)
#define SELB_STR 520     // bf16 sel row stride (pad, 520*2=1040=16*65)

typedef __attribute__((ext_vector_type(16))) __bf16 v16bf;
typedef __attribute__((ext_vector_type(8)))  __bf16 v8bf;
typedef __attribute__((ext_vector_type(8)))  float  v8f;

// LDS layout (dynamic shared mem), byte offsets:
#define SM_CAND_OFF 0                                   // u64 [16][512]  = 65536
#define SM_SELF_OFF (SM_CAND_OFF + 65536)               // f32 [16][516]  = 33024
#define SM_X_OFF    (SM_SELF_OFF + 33024)               // f32 [16][128]  = 8192
#define SM_HIST_OFF (SM_X_OFF + 8192)                   // u32 [16][256]  = 16384
#define SM_MISC_OFF (SM_HIST_OFF + 16384)               // u32 [64]       = 256
#define SM_SELB_OFF (SM_MISC_OFF + 256)                 // bf16 [16][520] = 16640
#define SM_KEYS_OFF (SM_SELB_OFF + 16640)               // u16 [16][4104] = 131328
#define SM_TOTAL    (SM_KEYS_OFF + 131328)              // = 271360 bytes (< 320KB WGP LDS)

__device__ __forceinline__ unsigned mono32(float f) {
  // monotone float->uint mapping (total order matches float <)
  unsigned u = __float_as_uint(f);
  return u ^ (unsigned)(((int)u >> 31) | (int)0x80000000);
}

// ---------------------------------------------------------------------------
// Prep: basis f32 -> bf16 copy (for WMMA B-operands) + per-anchor squared norm
// ---------------------------------------------------------------------------
__global__ void lcc_prep_kernel(const float* __restrict__ basis,
                                __bf16* __restrict__ basisB,
                                float* __restrict__ bn) {
  __shared__ float red[256];
  int a = blockIdx.x;
  int t = threadIdx.x;
  const float* row = basis + (size_t)a * Dsz;
  float v0 = row[t];
  float v1 = row[t + 256];
  __bf16* ob = basisB + (size_t)a * Dsz;
  ob[t]       = (__bf16)v0;
  ob[t + 256] = (__bf16)v1;
  red[t] = v0 * v0 + v1 * v1;
  __syncthreads();
  for (int s = 128; s > 0; s >>= 1) {
    if (t < s) red[t] += red[t + s];
    __syncthreads();
  }
  if (t == 0) bn[a] = red[0];
}

// ---------------------------------------------------------------------------
// Main: per 16 samples -> bf16-WMMA key GEMM (dual-tile, 2 independent WMMA
//       chains for XDL pipelining at 1-wave/SIMD occupancy), u16 histogram
//       prefilter, exact f32 refine, u64 bitonic sort, sparse output.
// ---------------------------------------------------------------------------
__global__ void __launch_bounds__(NTHREADS, 1)
lcc_main_kernel(const float* __restrict__ x,
                const int*   __restrict__ index,
                const float* __restrict__ basis,
                const __bf16* __restrict__ basisB,
                const float* __restrict__ bn,
                float* __restrict__ out) {
  extern __shared__ char smem[];
  unsigned long long* cand = (unsigned long long*)(smem + SM_CAND_OFF);
  float*    selF = (float*)(smem + SM_SELF_OFF);
  float*    xl   = (float*)(smem + SM_X_OFF);
  unsigned* hist = (unsigned*)(smem + SM_HIST_OFF);
  unsigned* hiBin = (unsigned*)(smem + SM_MISC_OFF);
  unsigned* cumB  = hiBin + 16;
  unsigned* thrU  = hiBin + 32;
  unsigned* cnt   = hiBin + 48;
  __bf16*   selB = (__bf16*)(smem + SM_SELB_OFF);
  unsigned short* keys = (unsigned short*)(smem + SM_KEYS_OFF);

  const int tid = threadIdx.x;
  const int b0  = blockIdx.x * ROWS;

  // ---- load selected anchors (f32 + bf16) and x rows into LDS ----
  for (int i = tid; i < ROWS * Dsz; i += NTHREADS) {
    int r = i >> 9;
    int d = i & (Dsz - 1);
    int ai = index[b0 + r];
    float v = basis[(size_t)ai * Dsz + d];
    selF[r * SELF_STR + d] = v;
    selB[r * SELB_STR + d] = (__bf16)v;
  }
  for (int i = tid; i < ROWS * Ssz; i += NTHREADS) {
    int r = i >> 7;
    int j = i & (Ssz - 1);
    xl[r * Ssz + j] = x[(size_t)(b0 + r) * Ssz + j];
  }
  if (tid < ROWS) cnt[tid] = 0;
  __syncthreads();

  const int lane = tid & 31;
  const int wave = tid >> 5;
  const int nl   = lane & 15;   // A-row (M) for lanes, and B-col (N) for lanes
  const int hi   = lane >> 4;   // lane half selects K sub-range per ISA layout

  for (int half = 0; half < 2; ++half) {
    const int cbase0 = half * HALFC;

    // ---- GEMM phase: keys[m][col] = bn[col] - 2 * <sel_m, basis_col> ----
    // Each wave processes a 16x32 column pair per iteration: two independent
    // accumulator chains keep the XDL pipe busy; A-frags are shared.
    for (int tp = wave; tp < HALFC / 32; tp += 4) {
      const int colg0 = cbase0 + tp * 32 + nl;
      const int colg1 = colg0 + 16;
      v8f c0 = {}, c1 = {};
      const __bf16* bcol0 = basisB + (size_t)colg0 * Dsz + hi * 16;
      const __bf16* bcol1 = basisB + (size_t)colg1 * Dsz + hi * 16;
      const __bf16* arow  = selB + nl * SELB_STR + hi * 16;
#pragma unroll
      for (int kk = 0; kk < Dsz; kk += 32) {
        // A 16x32 bf16 frag: lane holds row nl, K = kk+hi*16 .. +16 (contiguous)
        v8bf alo = *(const v8bf*)(arow + kk);
        v8bf ahi = *(const v8bf*)(arow + kk + 8);
        // B 32x16 bf16 frags: lane holds col nl, K = kk+hi*16 .. +16 (contiguous)
        v8bf b0lo = *(const v8bf*)(bcol0 + kk);
        v8bf b0hi = *(const v8bf*)(bcol0 + kk + 8);
        v8bf b1lo = *(const v8bf*)(bcol1 + kk);
        v8bf b1hi = *(const v8bf*)(bcol1 + kk + 8);
        v16bf av, bv0, bv1;
#pragma unroll
        for (int q = 0; q < 8; ++q) {
          av[q]  = alo[q];  av[8 + q]  = ahi[q];
          bv0[q] = b0lo[q]; bv0[8 + q] = b0hi[q];
          bv1[q] = b1lo[q]; bv1[8 + q] = b1hi[q];
        }
        c0 = __builtin_amdgcn_wmma_f32_16x16x32_bf16(false, av, false, bv0,
                                                     (short)0, c0, false, false);
        c1 = __builtin_amdgcn_wmma_f32_16x16x32_bf16(false, av, false, bv1,
                                                     (short)0, c1, false, false);
      }
      const float bnv0 = bn[colg0];
      const float bnv1 = bn[colg1];
      const int colL0 = tp * 32 + nl;    // column local to this half
      const int colL1 = colL0 + 16;
#pragma unroll
      for (int r8 = 0; r8 < 8; ++r8) {
        int m = r8 + 8 * hi;             // C/D layout: VGPR r, lanes 16-31 = rows 8..15
        float k0 = bnv0 - 2.0f * c0[r8];
        float k1 = bnv1 - 2.0f * c1[r8];
        keys[m * KSTRIDE + colL0] = (unsigned short)(mono32(k0) >> 16);
        keys[m * KSTRIDE + colL1] = (unsigned short)(mono32(k1) >> 16);
      }
    }
    __syncthreads();

    // ---- level-1 histogram on high byte of u16 keys ----
    for (int i = tid; i < ROWS * 256; i += NTHREADS) hist[i] = 0;
    __syncthreads();
    {
      int r = tid >> 3, t8 = tid & 7;
      for (int c2 = t8; c2 < HALFC; c2 += 8)
        atomicAdd(&hist[r * 256 + (keys[r * KSTRIDE + c2] >> 8)], 1u);
    }
    __syncthreads();
    if (tid < ROWS) {
      unsigned cum = 0; int bsel = 255;
      for (int bI = 0; bI < 256; ++bI) {
        unsigned h = hist[tid * 256 + bI];
        if (cum + h >= (unsigned)Ssz) { bsel = bI; break; }
        cum += h;
      }
      hiBin[tid] = (unsigned)bsel;
      cumB[tid]  = cum;
    }
    __syncthreads();

    // ---- level-2 histogram on low byte within selected high-byte bin ----
    for (int i = tid; i < ROWS * 256; i += NTHREADS) hist[i] = 0;
    __syncthreads();
    {
      int r = tid >> 3, t8 = tid & 7;
      unsigned hb = hiBin[r];
      for (int c2 = t8; c2 < HALFC; c2 += 8) {
        unsigned k = keys[r * KSTRIDE + c2];
        if ((k >> 8) == hb) atomicAdd(&hist[r * 256 + (k & 255u)], 1u);
      }
    }
    __syncthreads();
    if (tid < ROWS) {
      unsigned cum = cumB[tid]; int lsel = 255;
      for (int bI = 0; bI < 256; ++bI) {
        unsigned h = hist[tid * 256 + bI];
        if (cum + h >= (unsigned)Ssz) { lsel = bI; break; }
        cum += h;
      }
      unsigned v16 = (hiBin[tid] << 8) | (unsigned)lsel;
      // +1 bucket safety margin (bucket width ~8 >> bf16 key error ~0.2)
      thrU[tid] = (v16 >= 65535u) ? 65535u : (v16 + 1u);
    }
    __syncthreads();

    // ---- gather candidate columns (set only; exact order fixed later) ----
    {
      int r = tid >> 3, t8 = tid & 7;
      unsigned thr = thrU[r];
      for (int c2 = t8; c2 < HALFC; c2 += 8) {
        if ((unsigned)keys[r * KSTRIDE + c2] <= thr) {
          unsigned pos = atomicAdd(&cnt[r], 1u);
          if (pos < NC) cand[r * NC + pos] = (unsigned long long)(cbase0 + c2);
        }
      }
    }
    __syncthreads();
  }

  // ---- exact f32 refine of candidate keys; pack sortkey = mono32(key):col ----
  for (int i = tid; i < ROWS * NC; i += NTHREADS) {
    int r = i >> 9;
    int j = i & (NC - 1);
    unsigned n = cnt[r]; if (n > NC) n = NC;
    if ((unsigned)j < n) {
      unsigned col = (unsigned)cand[r * NC + j];
      const float4* bp = (const float4*)(basis + (size_t)col * Dsz);
      const float4* sp = (const float4*)(selF + r * SELF_STR);
      float4 acc = make_float4(0.f, 0.f, 0.f, 0.f);
      for (int k = 0; k < Dsz / 4; ++k) {
        float4 a4 = bp[k], s4 = sp[k];
        acc.x = fmaf(a4.x, s4.x, acc.x);
        acc.y = fmaf(a4.y, s4.y, acc.y);
        acc.z = fmaf(a4.z, s4.z, acc.z);
        acc.w = fmaf(a4.w, s4.w, acc.w);
      }
      float dot = (acc.x + acc.y) + (acc.z + acc.w);
      float key = bn[col] - 2.0f * dot;
      cand[r * NC + j] = ((unsigned long long)mono32(key) << 16) |
                         (unsigned long long)col;
    } else {
      cand[r * NC + j] = 0xFFFFFFFFFFFFFFFFull;   // +inf padding
    }
  }
  __syncthreads();

  // ---- bitonic sort 512 u64 per row, ascending (key, then col) ----
  {
    int r = tid >> 3, t8 = tid & 7;
    for (int k = 2; k <= NC; k <<= 1) {
      for (int j2 = k >> 1; j2 > 0; j2 >>= 1) {
        for (int i = t8; i < NC; i += 8) {
          int ixj = i ^ j2;
          if (ixj > i) {
            unsigned long long a = cand[r * NC + i];
            unsigned long long b = cand[r * NC + ixj];
            bool up = ((i & k) == 0);
            if ((a > b) == up) {
              cand[r * NC + i]   = b;
              cand[r * NC + ixj] = a;
            }
          }
        }
        __syncthreads();
      }
    }
  }

  // ---- out[b,:] = sum_j x[b,j] * basis[sorted_col_j,:]  (j-th nearest) ----
  for (int r = 0; r < ROWS; ++r) {
    float4 acc = make_float4(0.f, 0.f, 0.f, 0.f);
    for (int j = 0; j < Ssz; ++j) {
      unsigned long long e = cand[r * NC + j];
      unsigned col = (unsigned)(e & 0xFFFFull);
      float xv = xl[r * Ssz + j];
      if (col < (unsigned)Asz) {
        const float4* bp = (const float4*)(basis + (size_t)col * Dsz);
        float4 b4 = bp[tid];
        acc.x = fmaf(xv, b4.x, acc.x);
        acc.y = fmaf(xv, b4.y, acc.y);
        acc.z = fmaf(xv, b4.z, acc.z);
        acc.w = fmaf(xv, b4.w, acc.w);
      }
    }
    ((float4*)out)[(size_t)(b0 + r) * (Dsz / 4) + tid] = acc;
  }
}

// ---------------------------------------------------------------------------
extern "C" void kernel_launch(void* const* d_in, const int* in_sizes, int n_in,
                              void* d_out, int out_size, void* d_ws, size_t ws_size,
                              hipStream_t stream) {
  const float* x     = (const float*)d_in[0];
  const int*   index = (const int*)d_in[1];   // harness: integer -> const int*
  const float* basis = (const float*)d_in[2];
  float* out = (float*)d_out;

  // workspace: [bf16 basis copy: 8 MB][bn: 32 KB]
  __bf16* basisB = (__bf16*)d_ws;
  float*  bnArr  = (float*)((char*)d_ws + (size_t)Asz * Dsz * sizeof(__bf16));

  lcc_prep_kernel<<<Asz, 256, 0, stream>>>(basis, basisB, bnArr);

  // Not a stream op; idempotent and safe under graph capture.
  (void)hipFuncSetAttribute((const void*)lcc_main_kernel,
                            hipFuncAttributeMaxDynamicSharedMemorySize, SM_TOTAL);

  lcc_main_kernel<<<Bsz / ROWS, NTHREADS, SM_TOTAL, stream>>>(
      x, index, basis, basisB, bnArr, out);
}
